// PointTransformer_62878321214061
// MI455X (gfx1250) — matrix operand order repaired and checked by reading, not compiled, not run on hardware
//
#include <hip/hip_runtime.h>
#include <hip/hip_bf16.h>
#include <math.h>

// ---------------------------------------------------------------------------
// Point Transformer block for MI455X (gfx1250, wave32, WMMA + TDM).
// - All 128/512-deep GEMMs on v_wmma_f32_16x16x32_f16 (f16 in, f32 accum),
//   8 accumulators per wave so each A fragment feeds 8 WMMAs.
// - Neighbor kv gather done by the Tensor Data Mover in gather mode,
//   overlapped with the pos-MLP, completed with s_wait_tensorcnt.
// ---------------------------------------------------------------------------

typedef __attribute__((ext_vector_type(16))) _Float16 v16h;
typedef __attribute__((ext_vector_type(8)))  _Float16 v8h;
typedef __attribute__((ext_vector_type(8)))  float    v8f;
typedef __attribute__((ext_vector_type(4)))  unsigned int u32x4;
typedef __attribute__((ext_vector_type(8)))  int      i32x8;
typedef __attribute__((ext_vector_type(4)))  int      i32x4;

static constexpr int Bb = 4;
static constexpr int Nn = 4096;
static constexpr int Cc = 128;
static constexpr int Kk = 16;
static constexpr int FF = 512;
static constexpr int BN = Bb * Nn;          // 16384 rows

// Toolchain discriminator for the TDM builtin arity (see bridge doc):
// therock-10.0 headers ship amd_gfx1250_TDM.h and use the 6-arg builtin;
// ROCm 7.2 has the 5-arg form.
#if __has_include(<hip/amd_detail/amd_gfx1250_TDM.h>)
#define PT_TDM_6ARG 1
#else
#define PT_TDM_6ARG 0
#endif

__device__ inline void pt_tensor_load(u32x4 g0, i32x8 g1, i32x4 g2, i32x4 g3) {
#if PT_TDM_6ARG
  i32x8 gz = {0, 0, 0, 0, 0, 0, 0, 0};
  __builtin_amdgcn_tensor_load_to_lds(g0, g1, g2, g3, gz, 0);
#else
  __builtin_amdgcn_tensor_load_to_lds(g0, g1, g2, g3, 0);
#endif
}

// ---------------------------------------------------------------------------
// Weight packing: fragment-ready f16 B-matrix layout.
// B-frag (32x16, K x N): lane l covers col = l&15, kbase = (l>>4)*8,
// element i = W[ks*32 + (i>>3)*16 + kbase + (i&7)][col].  Packed so each lane's 16 halfs are
// contiguous: P[(((ks*NT + nt)*32) + lane)*16 + i].
// ---------------------------------------------------------------------------
__global__ __launch_bounds__(256) void pt_pack_w(const float* __restrict__ W,
                                                 _Float16* __restrict__ P,
                                                 int Kdim, int Nout) {
  int p = blockIdx.x * 256 + threadIdx.x;
  int total = Kdim * Nout;
  if (p >= total) return;
  int i    = p & 15;
  int lane = (p >> 4) & 31;
  int tile = p >> 9;                 // (ks, nt) pair, 512 halfs each
  int NT   = Nout >> 4;
  int nt   = tile % NT;
  int ks   = tile / NT;
  int k    = ks * 32 + ((i >> 3) << 4) + ((lane >> 4) << 3) + (i & 7);
  int n    = nt * 16 + (lane & 15);
  P[p] = (_Float16)W[(size_t)k * Nout + n];
}

// ---------------------------------------------------------------------------
// KNN: one thread per query point, LDS-tiled candidate scan, register top-16.
// ---------------------------------------------------------------------------
__global__ __launch_bounds__(256) void pt_knn(const float* __restrict__ pos,
                                              int* __restrict__ idxOut) {
  const int b  = blockIdx.y;
  const int nq = blockIdx.x * 256 + threadIdx.x;
  __shared__ float tile[256 * 3];
  const float* pb = pos + (size_t)b * Nn * 3;
  const float qx = pb[nq * 3 + 0];
  const float qy = pb[nq * 3 + 1];
  const float qz = pb[nq * 3 + 2];

  float dbest[16];
  int   ibest[16];
#pragma unroll
  for (int j = 0; j < 16; ++j) { dbest[j] = 3.4e38f; ibest[j] = 0; }

  for (int t0 = 0; t0 < Nn; t0 += 256) {
    __syncthreads();
    int tl = threadIdx.x;
    tile[tl * 3 + 0] = pb[(t0 + tl) * 3 + 0];
    tile[tl * 3 + 1] = pb[(t0 + tl) * 3 + 1];
    tile[tl * 3 + 2] = pb[(t0 + tl) * 3 + 2];
    __syncthreads();
    for (int j = 0; j < 256; ++j) {
      float dx = qx - tile[j * 3 + 0];
      float dy = qy - tile[j * 3 + 1];
      float dz = qz - tile[j * 3 + 2];
      float d  = dx * dx + dy * dy + dz * dz;
      if (d < dbest[15]) {
        dbest[15] = d; ibest[15] = t0 + j;
#pragma unroll
        for (int s = 15; s > 0; --s) {
          if (dbest[s] < dbest[s - 1]) {
            float td = dbest[s]; dbest[s] = dbest[s - 1]; dbest[s - 1] = td;
            int   ti = ibest[s]; ibest[s] = ibest[s - 1]; ibest[s - 1] = ti;
          }
        }
      }
    }
  }
#pragma unroll
  for (int j = 0; j < 16; ++j)
    idxOut[(size_t)(b * Nn + nq) * Kk + j] = ibest[j];
}

// ---------------------------------------------------------------------------
// WMMA GEMM with A reuse: one wave computes a 16-row x 128-col strip
// (8 n-tiles). A fragment is loaded/converted once per k-step and feeds
// 8 back-to-back WMMAs. grid = (M/16, Nout/128).
// ---------------------------------------------------------------------------
__global__ __launch_bounds__(32) void pt_gemm_multi(const float* __restrict__ A,
                                                    const _Float16* __restrict__ Wpk,
                                                    const float* __restrict__ bias,
                                                    const float* __restrict__ resid,
                                                    float* __restrict__ Out,
                                                    int Kdim, int Nout) {
  const int lane = threadIdx.x;
  const int mt   = blockIdx.x;
  const int ng   = blockIdx.y;       // group of 8 n-tiles (128 columns)
  const int NT   = Nout >> 4;
  const int rowA = mt * 16 + (lane & 15);
  const int kb   = (lane >> 4) * 8;
  const int cl   = lane & 15;

  v8f acc[8];
#pragma unroll
  for (int nt = 0; nt < 8; ++nt) {
    const float bv = bias ? bias[(ng * 8 + nt) * 16 + cl] : 0.0f;
#pragma unroll
    for (int r = 0; r < 8; ++r) acc[nt][r] = bv;
  }

  const int ksteps = Kdim >> 5;
  for (int ks = 0; ks < ksteps; ++ks) {
    const float* ap = A + (size_t)rowA * Kdim + ks * 32;
    __builtin_prefetch(ap + 32, 0, 0);
    v16h a;
#pragma unroll
    for (int i = 0; i < 8; ++i) {
      a[i]     = (_Float16)ap[kb + i];
      a[8 + i] = (_Float16)ap[16 + kb + i];
    }
#pragma unroll
    for (int nt = 0; nt < 8; ++nt) {
      v16h bfr = *(const v16h*)(Wpk +
                  ((size_t)(ks * NT + ng * 8 + nt) * 32 + lane) * 16);
      acc[nt] = __builtin_amdgcn_wmma_f32_16x16x32_f16(false, a, false, bfr,
                                                       (short)0, acc[nt],
                                                       false, false);
    }
  }

  const int mbase = mt * 16 + ((lane >> 4) << 3);
#pragma unroll
  for (int nt = 0; nt < 8; ++nt) {
    const int col = (ng * 8 + nt) * 16 + cl;
#pragma unroll
    for (int r = 0; r < 8; ++r) {
      size_t o = (size_t)(mbase + r) * Nout + col;
      float v = acc[nt][r];
      if (resid) v += resid[o];
      Out[o] = v;
    }
  }
}

// ---------------------------------------------------------------------------
// WMMA tile helper for the fused attention kernel: A is 16 x Kd f16 in LDS
// (row major), B is packed weights in global, returns one 16x16 f32 tile.
// ---------------------------------------------------------------------------
__device__ inline v8f pt_wmma_lds(const _Float16* __restrict__ Albs, int Kd,
                                  const _Float16* __restrict__ Wpk, int NT,
                                  int nt, float bval, int lane) {
  v8f acc;
#pragma unroll
  for (int r = 0; r < 8; ++r) acc[r] = bval;
  const int row = lane & 15;
  const int kb  = (lane >> 4) * 8;
  const int ksteps = Kd >> 5;
  for (int ks = 0; ks < ksteps; ++ks) {
    const _Float16* ap = Albs + row * Kd + ks * 32;
    v8h lo = *(const v8h*)(ap + kb);
    v8h hi = *(const v8h*)(ap + 16 + kb);
    v16h a;
#pragma unroll
    for (int i = 0; i < 8; ++i) { a[i] = lo[i]; a[8 + i] = hi[i]; }
    v16h bfr = *(const v16h*)(Wpk + ((size_t)(ks * NT + nt) * 32 + lane) * 16);
    acc = __builtin_amdgcn_wmma_f32_16x16x32_f16(false, a, false, bfr,
                                                 (short)0, acc, false, false);
  }
  return acc;
}

// ---------------------------------------------------------------------------
// Fused per-point attention: one wave per point; the 16 neighbors form one
// WMMA M-tile. The kv gather for the 16 neighbors is done by the TDM in
// gather mode (async, overlapped with the pos-MLP), then pos-MLP L2 and
// attn-MLP L1/L2 run on WMMA; softmax over K and aggregation in VALU.
// 2 waves/block, per-wave LDS region of WAVE_LDS bytes.
// ---------------------------------------------------------------------------
static constexpr int OFF_QROW = 0;        // 128 f32   (512 B)
static constexpr int OFF_PD   = 512;      // 16x4 f32  (256 B)
static constexpr int OFF_PE   = 768;      // 16x128 f32 (8192 B)
static constexpr int OFF_VPE  = 8960;     // 16x128 f32 (8192 B)
static constexpr int OFF_FB0  = 17152;    // 16x128 f16 (4096 B)
static constexpr int OFF_FB1  = 21248;    // 16x128 f16 (4096 B)
static constexpr int OFF_NID  = 25344;    // 16 int    (64 B)
static constexpr int OFF_KV   = 25600;    // 16x256 f32 (16384 B) TDM dest
static constexpr int WAVE_LDS = 41984;    // bytes per wave (multiple of 256)

__global__ __launch_bounds__(64) void pt_attn_fused(
    const float* __restrict__ pos, const float* __restrict__ q,
    const float* __restrict__ kv, const int* __restrict__ nidG,
    const _Float16* __restrict__ pWp2, const _Float16* __restrict__ pWa1,
    const _Float16* __restrict__ pWa2, const float* __restrict__ Wp1,
    const float* __restrict__ bp1, const float* __restrict__ bp2,
    const float* __restrict__ ba1, const float* __restrict__ ba2,
    float* __restrict__ agg) {
  extern __shared__ char smem[];
  const int wave = threadIdx.x >> 5;
  const int lane = threadIdx.x & 31;
  const int point = blockIdx.x * 2 + wave;  // global row, = b*Nn + n
  const int b = point >> 12;                // Nn == 4096

  char* wl = smem + wave * WAVE_LDS;
  float*    qrow  = (float*)(wl + OFF_QROW);
  float*    pd    = (float*)(wl + OFF_PD);
  float*    pe    = (float*)(wl + OFF_PE);
  float*    vpe   = (float*)(wl + OFF_VPE);
  _Float16* fb0   = (_Float16*)(wl + OFF_FB0);
  _Float16* fb1   = (_Float16*)(wl + OFF_FB1);
  int*      nid   = (int*)(wl + OFF_NID);
  float*    kvlds = (float*)(wl + OFF_KV);

  // --- neighbor ids + q row --------------------------------------------------
  if (lane < 16) nid[lane] = nidG[(size_t)point * Kk + lane];
  for (int c = lane; c < 128; c += 32) qrow[c] = q[(size_t)point * Cc + c];
  __syncthreads();

  // --- TDM gather of the 16 neighbor kv rows (async, overlapped) -------------
  {
    unsigned iw[8];
#pragma unroll
    for (int w2 = 0; w2 < 8; ++w2) {
      unsigned lo2 = (unsigned)nid[w2 * 2 + 0] & 0xFFFFu;
      unsigned hi2 = (unsigned)nid[w2 * 2 + 1] & 0xFFFFu;
      iw[w2] = lo2 | (hi2 << 16);
    }
    const unsigned long long ga =
        (unsigned long long)(size_t)(kv + (size_t)b * Nn * (2 * Cc));
    const unsigned ldsOff = (unsigned)(wave * WAVE_LDS + OFF_KV);

    u32x4 g0;
    g0[0] = 0x80000001u;                       // gather_mode=1, 16b idx, count=1
    g0[1] = ldsOff;                            // lds_addr
    g0[2] = (unsigned)(ga & 0xFFFFFFFFu);      // global_addr[31:0]
    g0[3] = (unsigned)((ga >> 32) & 0x01FFFFFFu) | 0x80000000u;  // type=2

    i32x8 g1;
    g1[0] = (2 << 16);            // workgroup_mask=0, data_size=2 (4B)
    g1[1] = (int)(256u << 16);    // tensor_dim0[15:0]=256 (kv row, f32 units)
    g1[2] = (int)(4096u << 16);   // tensor_dim0 hi=0, tensor_dim1[15:0]=4096
    g1[3] = (int)(256u << 16);    // tensor_dim1 hi=0, tile_dim0=256
    g1[4] = 16;                   // tile_dim1 = 16 valid indices, tile_dim2=0
    g1[5] = 256;                  // tensor_dim0_stride[31:0]=256
    g1[6] = 0;
    g1[7] = 0;

    i32x4 g2, g3;
#pragma unroll
    for (int w2 = 0; w2 < 4; ++w2) { g2[w2] = (int)iw[w2]; g3[w2] = (int)iw[4 + w2]; }
    pt_tensor_load(g0, g1, g2, g3);
  }

  // --- relative positions ----------------------------------------------------
  if (lane < 16) {
    int nb = nid[lane];
    const float* pq = pos + (size_t)point * 3;
    const float* pn = pos + (size_t)(b * Nn + nb) * 3;
    pd[lane * 4 + 0] = pq[0] - pn[0];
    pd[lane * 4 + 1] = pq[1] - pn[1];
    pd[lane * 4 + 2] = pq[2] - pn[2];
  }
  __syncthreads();

  // --- pos MLP layer 1 (K=3 -> VALU) + relu, to f16 --------------------------
  for (int c = lane; c < 128; c += 32) {
    float w0 = Wp1[c], w1 = Wp1[128 + c], w2 = Wp1[256 + c], bb = bp1[c];
#pragma unroll
    for (int row = 0; row < 16; ++row) {
      float h = pd[row * 4 + 0] * w0 + pd[row * 4 + 1] * w1 +
                pd[row * 4 + 2] * w2 + bb;
      fb0[row * 128 + c] = (_Float16)fmaxf(h, 0.0f);
    }
  }
  __syncthreads();

  // --- pos MLP layer 2 (WMMA) -> pos_enc (f32 in LDS) ------------------------
  const int mb = (lane >> 4) << 3;
  const int cl = lane & 15;
  for (int nt = 0; nt < 8; ++nt) {
    v8f acc = pt_wmma_lds(fb0, 128, pWp2, 8, nt, bp2[nt * 16 + cl], lane);
#pragma unroll
    for (int r = 0; r < 8; ++r) pe[(mb + r) * 128 + nt * 16 + cl] = acc[r];
  }
  __syncthreads();

  // --- TDM gather done? then rel = k_n - q + pos_enc, vpe = v_n + pos_enc ----
  __builtin_amdgcn_s_wait_tensorcnt(0);
  for (int c = lane; c < 128; c += 32) {
    float qv = qrow[c];
#pragma unroll
    for (int row = 0; row < 16; ++row) {
      float pev = pe[row * 128 + c];
      float kvk = kvlds[row * 256 + c];
      float kvv = kvlds[row * 256 + 128 + c];
      fb1[row * 128 + c] = (_Float16)(kvk - qv + pev);
      vpe[row * 128 + c] = kvv + pev;
    }
  }
  __syncthreads();

  // --- attn MLP layer 1 (WMMA) + relu, to f16 --------------------------------
  for (int nt = 0; nt < 8; ++nt) {
    v8f acc = pt_wmma_lds(fb1, 128, pWa1, 8, nt, ba1[nt * 16 + cl], lane);
#pragma unroll
    for (int r = 0; r < 8; ++r)
      fb0[(mb + r) * 128 + nt * 16 + cl] = (_Float16)fmaxf(acc[r], 0.0f);
  }
  __syncthreads();

  // --- attn MLP layer 2 (WMMA) -> logits (reuse pe) --------------------------
  for (int nt = 0; nt < 8; ++nt) {
    v8f acc = pt_wmma_lds(fb0, 128, pWa2, 8, nt, ba2[nt * 16 + cl], lane);
#pragma unroll
    for (int r = 0; r < 8; ++r) pe[(mb + r) * 128 + nt * 16 + cl] = acc[r];
  }
  __syncthreads();

  // --- softmax over K (rows) per channel, aggregate with vpe -----------------
  for (int c = lane; c < 128; c += 32) {
    float mx = -3.4e38f;
#pragma unroll
    for (int row = 0; row < 16; ++row) mx = fmaxf(mx, pe[row * 128 + c]);
    float s = 0.0f;
#pragma unroll
    for (int row = 0; row < 16; ++row) {
      float e = __expf(pe[row * 128 + c] - mx);
      pe[row * 128 + c] = e;
      s += e;
    }
    float inv = 1.0f / s;
    float a = 0.0f;
#pragma unroll
    for (int row = 0; row < 16; ++row)
      a += pe[row * 128 + c] * vpe[row * 128 + c];
    agg[(size_t)point * Cc + c] = a * inv;
  }
}

// ---------------------------------------------------------------------------
// LayerNorm(512) + exact GELU, in place. One wave per row.
// ---------------------------------------------------------------------------
__global__ __launch_bounds__(32) void pt_ln_gelu_512(float* __restrict__ t,
                                                     const float* __restrict__ g,
                                                     const float* __restrict__ bta) {
  const int row = blockIdx.x;
  const int lane = threadIdx.x;
  float* r = t + (size_t)row * FF;
  float vals[16];
  float s = 0.0f;
#pragma unroll
  for (int i = 0; i < 16; ++i) { vals[i] = r[lane + i * 32]; s += vals[i]; }
#pragma unroll
  for (int off = 16; off > 0; off >>= 1) s += __shfl_xor(s, off, 32);
  const float mean = s * (1.0f / FF);
  float v = 0.0f;
#pragma unroll
  for (int i = 0; i < 16; ++i) { float d = vals[i] - mean; v += d * d; }
#pragma unroll
  for (int off = 16; off > 0; off >>= 1) v += __shfl_xor(v, off, 32);
  const float rstd = rsqrtf(v * (1.0f / FF) + 1e-5f);
#pragma unroll
  for (int i = 0; i < 16; ++i) {
    int c = lane + i * 32;
    float h = (vals[i] - mean) * rstd * g[c] + bta[c];
    h = 0.5f * h * (1.0f + erff(h * 0.70710678118654752f));
    r[c] = h;
  }
}

// ---------------------------------------------------------------------------
// LayerNorm(128) + residual: out = x1 + LN(t2). One wave per row.
// ---------------------------------------------------------------------------
__global__ __launch_bounds__(32) void pt_ln_res_128(const float* __restrict__ t2,
                                                    const float* __restrict__ x1,
                                                    const float* __restrict__ g,
                                                    const float* __restrict__ bta,
                                                    float* __restrict__ out) {
  const int row = blockIdx.x;
  const int lane = threadIdx.x;
  const float* r = t2 + (size_t)row * Cc;
  float vals[4];
  float s = 0.0f;
#pragma unroll
  for (int i = 0; i < 4; ++i) { vals[i] = r[lane + i * 32]; s += vals[i]; }
#pragma unroll
  for (int off = 16; off > 0; off >>= 1) s += __shfl_xor(s, off, 32);
  const float mean = s * (1.0f / Cc);
  float v = 0.0f;
#pragma unroll
  for (int i = 0; i < 4; ++i) { float d = vals[i] - mean; v += d * d; }
#pragma unroll
  for (int off = 16; off > 0; off >>= 1) v += __shfl_xor(v, off, 32);
  const float rstd = rsqrtf(v * (1.0f / Cc) + 1e-5f);
#pragma unroll
  for (int i = 0; i < 4; ++i) {
    int c = lane + i * 32;
    float h = (vals[i] - mean) * rstd * g[c] + bta[c];
    out[(size_t)row * Cc + c] = x1[(size_t)row * Cc + c] + h;
  }
}

// ---------------------------------------------------------------------------
// Host-side orchestration.
// ---------------------------------------------------------------------------
extern "C" void kernel_launch(void* const* d_in, const int* in_sizes, int n_in,
                              void* d_out, int out_size, void* d_ws,
                              size_t ws_size, hipStream_t stream) {
  (void)in_sizes; (void)n_in; (void)out_size; (void)ws_size;
  const float* x    = (const float*)d_in[0];
  const float* pos  = (const float*)d_in[1];
  const float* Wq   = (const float*)d_in[2];
  const float* bq   = (const float*)d_in[3];
  const float* Wkv  = (const float*)d_in[4];
  const float* bkv  = (const float*)d_in[5];
  const float* Wp1  = (const float*)d_in[6];
  const float* bp1  = (const float*)d_in[7];
  const float* Wp2  = (const float*)d_in[8];
  const float* bp2  = (const float*)d_in[9];
  const float* Wa1  = (const float*)d_in[10];
  const float* ba1  = (const float*)d_in[11];
  const float* Wa2  = (const float*)d_in[12];
  const float* ba2  = (const float*)d_in[13];
  const float* Wo   = (const float*)d_in[14];
  const float* bo   = (const float*)d_in[15];
  const float* Wf1  = (const float*)d_in[16];
  const float* bf1  = (const float*)d_in[17];
  const float* ln1g = (const float*)d_in[18];
  const float* ln1b = (const float*)d_in[19];
  const float* Wf2  = (const float*)d_in[20];
  const float* bf2  = (const float*)d_in[21];
  const float* ln2g = (const float*)d_in[22];
  const float* ln2b = (const float*)d_in[23];
  float* out = (float*)d_out;

  // workspace partition (all 256B aligned)
  char* p = (char*)d_ws;
  auto alloc = [&](size_t bytes) {
    char* r = p;
    p += (bytes + 255) & ~(size_t)255;
    return r;
  };
  int*      idx  = (int*)alloc((size_t)BN * Kk * sizeof(int));
  float*    qb   = (float*)alloc((size_t)BN * Cc * 4);
  float*    kvb  = (float*)alloc((size_t)BN * 2 * Cc * 4);
  float*    aggb = (float*)alloc((size_t)BN * Cc * 4);
  float*    x1b  = (float*)alloc((size_t)BN * Cc * 4);
  float*    t1b  = (float*)alloc((size_t)BN * FF * 4);
  float*    t2b  = (float*)alloc((size_t)BN * Cc * 4);
  _Float16* pWq  = (_Float16*)alloc((size_t)Cc * Cc * 2);
  _Float16* pWkv = (_Float16*)alloc((size_t)Cc * 2 * Cc * 2);
  _Float16* pWp2 = (_Float16*)alloc((size_t)Cc * Cc * 2);
  _Float16* pWa1 = (_Float16*)alloc((size_t)Cc * Cc * 2);
  _Float16* pWa2 = (_Float16*)alloc((size_t)Cc * Cc * 2);
  _Float16* pWo  = (_Float16*)alloc((size_t)Cc * Cc * 2);
  _Float16* pWf1 = (_Float16*)alloc((size_t)Cc * FF * 2);
  _Float16* pWf2 = (_Float16*)alloc((size_t)FF * Cc * 2);

  // pack weights into fragment-ready f16 layout
  pt_pack_w<<<(Cc * Cc + 255) / 256, 256, 0, stream>>>(Wq, pWq, Cc, Cc);
  pt_pack_w<<<(Cc * 2 * Cc + 255) / 256, 256, 0, stream>>>(Wkv, pWkv, Cc, 2 * Cc);
  pt_pack_w<<<(Cc * Cc + 255) / 256, 256, 0, stream>>>(Wp2, pWp2, Cc, Cc);
  pt_pack_w<<<(Cc * Cc + 255) / 256, 256, 0, stream>>>(Wa1, pWa1, Cc, Cc);
  pt_pack_w<<<(Cc * Cc + 255) / 256, 256, 0, stream>>>(Wa2, pWa2, Cc, Cc);
  pt_pack_w<<<(Cc * Cc + 255) / 256, 256, 0, stream>>>(Wo, pWo, Cc, Cc);
  pt_pack_w<<<(Cc * FF + 255) / 256, 256, 0, stream>>>(Wf1, pWf1, Cc, FF);
  pt_pack_w<<<(FF * Cc + 255) / 256, 256, 0, stream>>>(Wf2, pWf2, FF, Cc);

  // knn
  pt_knn<<<dim3(Nn / 256, Bb), 256, 0, stream>>>(pos, idx);

  // q / kv projections (A reused across 8 n-tiles per wave)
  pt_gemm_multi<<<dim3(BN / 16, 1), 32, 0, stream>>>(x, pWq, bq, nullptr, qb,
                                                     Cc, Cc);
  pt_gemm_multi<<<dim3(BN / 16, 2), 32, 0, stream>>>(x, pWkv, bkv, nullptr,
                                                     kvb, Cc, 2 * Cc);

  // fused neighbor attention (one wave per point, TDM gather for kv)
  pt_attn_fused<<<BN / 2, 64, 2 * WAVE_LDS, stream>>>(
      pos, qb, kvb, idx, pWp2, pWa1, pWa2, Wp1, bp1, bp2, ba1, ba2, aggb);

  // x1 = x + agg @ Wo + bo
  pt_gemm_multi<<<dim3(BN / 16, 1), 32, 0, stream>>>(aggb, pWo, bo, x, x1b,
                                                     Cc, Cc);
  // t1 = x1 @ Wf1 + bf1 ; h1 = gelu(LN(t1)) in place
  pt_gemm_multi<<<dim3(BN / 16, 4), 32, 0, stream>>>(x1b, pWf1, bf1, nullptr,
                                                     t1b, Cc, FF);
  pt_ln_gelu_512<<<BN, 32, 0, stream>>>(t1b, ln1g, ln1b);
  // t2 = h1 @ Wf2 + bf2 ; out = x1 + LN(t2)
  pt_gemm_multi<<<dim3(BN / 16, 1), 32, 0, stream>>>(t1b, pWf2, bf2, nullptr,
                                                     t2b, FF, Cc);
  pt_ln_res_128<<<BN, 32, 0, stream>>>(t2b, x1b, ln2g, ln2b, out);
}
